// modi_cgcnn_46248207843560
// MI455X (gfx1250) — matrix operand order repaired and checked by reading, not compiled, not run on hardware
//
// CGCNN conv layer for MI455X (gfx1250), wave32 + WMMA bf16.
// Bandwidth-bound design: single big bf16 WMMA GEMM (1.2M x 192 @ 192 x 128),
// one-pass sum/sumsq segment stats via LDS ds_add_f32 atomics, global f32
// atomics for edge->node scatter. Workspace requirement: ~641 MB.
#include <hip/hip_runtime.h>
#include <math.h>

#define DEVINL __device__ __forceinline__

// ---------------- problem sizes ----------------
constexpr int kNodes = 50000;
constexpr int kEdges = 1200000;
constexpr int kCrys  = 256;
constexpr int AF = 64, NF = 64, NR = 16;
constexpr int KDIM = 2 * AF + NF;   // 192
constexpr int NOUT = 2 * AF;        // 128
constexpr int NTILES = kEdges / 16; // 75000 (exact)
constexpr float kInvSqrt2 = 0.70710678118654752440f;
constexpr float kEps = 1e-5f;

// ---------------- vector types ----------------
typedef __attribute__((ext_vector_type(16))) __bf16 v16bf;
typedef __attribute__((ext_vector_type(8)))  __bf16 bf16x8;
typedef __attribute__((ext_vector_type(2)))  __bf16 bf16x2;
typedef __attribute__((ext_vector_type(8)))  float  v8f;

// ---------------- workspace layout (float offsets) ----------------
constexpr size_t O_GATED    = 0;                                     // kEdges*128
constexpr size_t O_ESUM     = (size_t)kEdges * NOUT;                 // zeroed region begins
constexpr size_t O_ESUMSQ   = O_ESUM    + (size_t)kCrys * NOUT;
constexpr size_t O_ECNT     = O_ESUMSQ  + (size_t)kCrys * NOUT;
constexpr size_t O_NODESUM  = O_ECNT    + kCrys;
constexpr size_t O_NODECNT  = O_NODESUM + (size_t)kNodes * AF;
constexpr size_t O_NSUM     = O_NODECNT + kNodes;
constexpr size_t O_NSUMSQ   = O_NSUM    + (size_t)kCrys * AF;
constexpr size_t O_NCNT     = O_NSUMSQ  + (size_t)kCrys * AF;
constexpr size_t O_ZERO_BEG = O_ESUM;
constexpr size_t O_ZERO_END = O_NCNT    + kCrys;
constexpr size_t O_EMEAN    = O_ZERO_END;
constexpr size_t O_EINV     = O_EMEAN   + (size_t)kCrys * NOUT;
constexpr size_t O_NS       = O_EINV    + (size_t)kCrys * NOUT;
constexpr size_t O_NMEAN    = O_NS      + (size_t)kNodes * AF;
constexpr size_t O_NINV     = O_NMEAN   + (size_t)kCrys * AF;

// ---------------- atomic helpers (native f32 add paths) ----------------
DEVINL void atomAddShared(float* p, float v) {
  __hip_atomic_fetch_add(p, v, __ATOMIC_RELAXED, __HIP_MEMORY_SCOPE_WORKGROUP);
}
DEVINL void atomAddGlobal(float* p, float v) {
  __hip_atomic_fetch_add(p, v, __ATOMIC_RELAXED, __HIP_MEMORY_SCOPE_AGENT);
}

// ---------------- 0: zero accumulators (ws is poisoned 0xAA) ----------------
__global__ void k_zero(float* __restrict__ p, size_t n) {
  size_t i  = (size_t)blockIdx.x * blockDim.x + threadIdx.x;
  size_t st = (size_t)gridDim.x * blockDim.x;
  for (; i < n; i += st) p[i] = 0.0f;
}

// ---------------- 1: edge GEMM via WMMA bf16 ----------------
// One 16-edge tile per wave. A(16x192) staged to LDS (bf16, row-major):
//   cols [0,64)   = atom_fea[idx0]
//   cols [64,128) = atom_fea[idx1]
//   cols [128,192)= edge * (rbf @ W_rbf)    (rbf GEMM done in f32 VALU)
// Then 8 N-subtiles x 6 K-steps of v_wmma_f32_16x16x32_bf16 against
// W_full^T resident in LDS. Output: gated (f32, row-major [edge][128]).
__global__ __launch_bounds__(64) void k_edge_gemm(
    const float* __restrict__ atom_fea, const float* __restrict__ edge,
    const float* __restrict__ rbf, const int* __restrict__ nbr_idx,
    const float* __restrict__ W_rbf, const float* __restrict__ W_full,
    float* __restrict__ gated)
{
  __shared__ alignas(16) __bf16 sWt[NOUT * KDIM];   // 48 KB: Wt[n][k] = W_full[k][n]
  __shared__ alignas(16) __bf16 sA[2][16 * KDIM];   // 12 KB: one tile per wave

  const int tid  = threadIdx.x;
  const int wave = tid >> 5;
  const int lane = tid & 31;

  // stage W_full^T as bf16 (L2-hot; amortized over many tiles per block)
  for (int i = tid; i < NOUT * KDIM; i += 64) {
    const int n = i / KDIM, k = i - n * KDIM;
    sWt[i] = (__bf16)W_full[k * NOUT + n];
  }
  __syncthreads();

  // per-lane W_rbf column pair (features 2*lane, 2*lane+1), kept in VGPRs
  const int f0 = lane * 2;
  float wr0[NR], wr1[NR];
#pragma unroll
  for (int r = 0; r < NR; ++r) {
    wr0[r] = W_rbf[r * NF + f0];
    wr1[r] = W_rbf[r * NF + f0 + 1];
  }

  __bf16* aT = sA[wave];
  const int m    = lane & 15;   // A row / B column / C column within subtile
  const int half = lane >> 4;

  for (int tile = blockIdx.x * 2 + wave; tile < NTILES; tile += gridDim.x * 2) {
    const int e0 = tile * 16;

    // ---- stage A tile (each lane supplies its 2-feature slice of each row) ----
    for (int r = 0; r < 16; ++r) {
      const int e  = e0 + r;
      const int i0 = nbr_idx[e * 2 + 0];
      const int i1 = nbr_idx[e * 2 + 1];
      const float2 a0 = *(const float2*)(atom_fea + (size_t)i0 * AF + f0);
      const float2 a1 = *(const float2*)(atom_fea + (size_t)i1 * AF + f0);
      const float* rb = rbf + (size_t)e * NR;
      float h0 = 0.0f, h1 = 0.0f;
#pragma unroll
      for (int q = 0; q < NR; ++q) {
        const float rv = rb[q];
        h0 = fmaf(rv, wr0[q], h0);
        h1 = fmaf(rv, wr1[q], h1);
      }
      const float2 ev = *(const float2*)(edge + (size_t)e * NF + f0);
      __bf16* rowp = aT + r * KDIM;
      bf16x2 p;
      p[0] = (__bf16)a0.x;        p[1] = (__bf16)a0.y;        *(bf16x2*)(rowp + f0)          = p;
      p[0] = (__bf16)a1.x;        p[1] = (__bf16)a1.y;        *(bf16x2*)(rowp + AF + f0)     = p;
      p[0] = (__bf16)(ev.x * h0); p[1] = (__bf16)(ev.y * h1); *(bf16x2*)(rowp + 2 * AF + f0) = p;
    }
    __builtin_amdgcn_fence(__ATOMIC_ACQ_REL, "wavefront");

    // ---- 16x192 @ 192x128 : 6 K-steps x 8 N-subtiles of WMMA bf16 ----
    v8f acc[8] = {};
#pragma unroll
    for (int kk = 0; kk < 6; ++kk) {
      // A fragment: lanes 0-15 take K {0..7,16..23}, lanes 16-31 take {8..15,24..31}
      const __bf16* ap = aT + m * KDIM + kk * 32 + half * 8;
      const bf16x8 alo = *(const bf16x8*)ap;
      const bf16x8 ahi = *(const bf16x8*)(ap + 16);
      v16bf a;
#pragma unroll
      for (int i = 0; i < 8; ++i) { a[i] = alo[i]; a[i + 8] = ahi[i]; }
#pragma unroll
      for (int n = 0; n < 8; ++n) {
        // B fragment: column N = lane%16, K {0..7,16..23} (lanes 0-15) / {8..15,24..31} (lanes 16-31)
        const __bf16* bp = sWt + (size_t)(n * 16 + m) * KDIM + kk * 32 + half * 8;
        const bf16x8 blo = *(const bf16x8*)bp;
        const bf16x8 bhi = *(const bf16x8*)(bp + 16);
        v16bf b;
#pragma unroll
        for (int i = 0; i < 8; ++i) { b[i] = blo[i]; b[i + 8] = bhi[i]; }
        acc[n] = __builtin_amdgcn_wmma_f32_16x16x32_bf16(
            false, a, false, b, (short)0, acc[n], false, false);
      }
    }
    __builtin_amdgcn_fence(__ATOMIC_ACQ_REL, "wavefront");

    // ---- store C: lane holds rows M = r + 8*half, col N = n*16 + (lane&15) ----
#pragma unroll
    for (int n = 0; n < 8; ++n) {
      float* gp = gated + (size_t)(e0 + 8 * half) * NOUT + n * 16 + m;
#pragma unroll
      for (int r = 0; r < 8; ++r) gp[(size_t)r * NOUT] = acc[n][r];
    }
  }
}

// ---------------- 2: edge crystal stats (sum & sumsq), feature-chunked ----------------
__global__ __launch_bounds__(256) void k_edge_stats(
    const float* __restrict__ gated, const int* __restrict__ cidx,
    float* __restrict__ esum, float* __restrict__ esumsq)
{
  __shared__ float ssum[kCrys * 32];  // 32 KB
  __shared__ float ssq [kCrys * 32];  // 32 KB
  const int fbase = blockIdx.y * 32;
  for (int i = threadIdx.x; i < kCrys * 32; i += 256) { ssum[i] = 0.0f; ssq[i] = 0.0f; }
  __syncthreads();
  for (int e = blockIdx.x * 256 + threadIdx.x; e < kEdges; e += gridDim.x * 256) {
    const int c = cidx[e];
    const float4* row = (const float4*)(gated + (size_t)e * NOUT + fbase);
    float* bs = ssum + c * 32;
    float* bq = ssq  + c * 32;
#pragma unroll
    for (int q = 0; q < 8; ++q) {
      const float4 v = row[q];
      atomAddShared(bs + q * 4 + 0, v.x); atomAddShared(bq + q * 4 + 0, v.x * v.x);
      atomAddShared(bs + q * 4 + 1, v.y); atomAddShared(bq + q * 4 + 1, v.y * v.y);
      atomAddShared(bs + q * 4 + 2, v.z); atomAddShared(bq + q * 4 + 2, v.z * v.z);
      atomAddShared(bs + q * 4 + 3, v.w); atomAddShared(bq + q * 4 + 3, v.w * v.w);
    }
  }
  __syncthreads();
  for (int i = threadIdx.x; i < kCrys * 32; i += 256) {
    const int c = i >> 5, f = i & 31;
    atomAddGlobal(esum   + (size_t)c * NOUT + fbase + f, ssum[i]);
    atomAddGlobal(esumsq + (size_t)c * NOUT + fbase + f, ssq[i]);
  }
}

// ---------------- 2b: edges per crystal ----------------
__global__ __launch_bounds__(256) void k_edge_cnt(
    const int* __restrict__ cidx, float* __restrict__ ecnt)
{
  __shared__ float scnt[kCrys];
  for (int i = threadIdx.x; i < kCrys; i += 256) scnt[i] = 0.0f;
  __syncthreads();
  for (int e = blockIdx.x * 256 + threadIdx.x; e < kEdges; e += gridDim.x * 256)
    atomAddShared(&scnt[cidx[e]], 1.0f);
  __syncthreads();
  for (int i = threadIdx.x; i < kCrys; i += 256) atomAddGlobal(ecnt + i, scnt[i]);
}

// ---------------- 3: finalize mean / inv_std (shared by edge & node stats) ----------------
__global__ void k_finalize(const float* __restrict__ sum, const float* __restrict__ sumsq,
                           const float* __restrict__ cnt, float* __restrict__ mean,
                           float* __restrict__ inv, int nFeat, int total)
{
  const int i = blockIdx.x * blockDim.x + threadIdx.x;
  if (i >= total) return;
  const float ct  = fmaxf(cnt[i / nFeat], 1.0f);
  const float m   = sum[i] / ct;
  const float var = fmaxf(sumsq[i] / ct - m * m, 0.0f);
  mean[i] = m;
  inv[i]  = rsqrtf(var + kEps);
}

// ---------------- 4: normalize, gate, scatter messages to nodes ----------------
__global__ __launch_bounds__(256) void k_edge_msg(
    const float* __restrict__ gated, const int* __restrict__ cidx,
    const int* __restrict__ nbr_idx, const float* __restrict__ emean,
    const float* __restrict__ einv, const float* __restrict__ g1,
    const float* __restrict__ b1, const float* __restrict__ W_mask,
    float* __restrict__ node_sum, float* __restrict__ node_cnt)
{
  __shared__ float sW[AF], sG[NOUT], sB[NOUT];
  for (int i = threadIdx.x; i < AF;   i += 256) sW[i] = W_mask[i];
  for (int i = threadIdx.x; i < NOUT; i += 256) { sG[i] = g1[i]; sB[i] = b1[i]; }
  __syncthreads();
  for (int e = blockIdx.x * 256 + threadIdx.x; e < kEdges; e += gridDim.x * 256) {
    const int c = cidx[e];
    const float* row = gated + (size_t)e * NOUT;
    const float* mu  = emean + (size_t)c * NOUT;
    const float* iv  = einv  + (size_t)c * NOUT;
    float dot = 0.0f;
#pragma unroll
    for (int f = 0; f < AF; ++f) {
      const float xn = (row[f] - mu[f]) * iv[f] * sG[f] + sB[f];
      dot = fmaf(xn, sW[f], dot);
    }
    const float s  = 1.0f / (1.0f + __expf(-dot));
    const int   n0 = nbr_idx[e * 2 + 0];
    float* np = node_sum + (size_t)n0 * AF;
#pragma unroll
    for (int f = 0; f < AF; ++f) {
      const float xn = (row[AF + f] - mu[AF + f]) * iv[AF + f] * sG[AF + f] + sB[AF + f];
      atomAddGlobal(np + f, s * fmaxf(xn, 0.0f));
    }
    atomAddGlobal(node_cnt + n0, 1.0f);
  }
}

// ---------------- 5: node mean-pool + node crystal stats ----------------
__global__ __launch_bounds__(256) void k_node_reduce(
    const float* __restrict__ node_sum, const float* __restrict__ node_cnt,
    const int* __restrict__ caidx, float* __restrict__ ns,
    float* __restrict__ nsum, float* __restrict__ nsumsq, float* __restrict__ ncnt)
{
  for (int n = blockIdx.x * 256 + threadIdx.x; n < kNodes; n += gridDim.x * 256) {
    const float inv = 1.0f / fmaxf(node_cnt[n], 1.0f);
    const int   c   = caidx[n];
    atomAddGlobal(ncnt + c, 1.0f);
    const float* sp = node_sum + (size_t)n * AF;
    float* op = ns     + (size_t)n * AF;
    float* cs = nsum   + (size_t)c * AF;
    float* cq = nsumsq + (size_t)c * AF;
#pragma unroll 8
    for (int f = 0; f < AF; ++f) {
      const float v = sp[f] * inv;
      op[f] = v;
      atomAddGlobal(cs + f, v);
      atomAddGlobal(cq + f, v * v);
    }
  }
}

// ---------------- 7: node norm + residual MLPs + output ----------------
DEVINL void mlp_block(float x[AF], const float* __restrict__ Wa,
                      const float* __restrict__ Wb)
{
  float h[32];
#pragma unroll
  for (int j = 0; j < 32; ++j) h[j] = 0.0f;
#pragma unroll
  for (int i = 0; i < AF; ++i) {
    const float xv = x[i];
    const float4* wrow = (const float4*)(Wa + i * 32);
#pragma unroll
    for (int q = 0; q < 8; ++q) {
      const float4 w = wrow[q];
      h[q * 4 + 0] = fmaf(xv, w.x, h[q * 4 + 0]);
      h[q * 4 + 1] = fmaf(xv, w.y, h[q * 4 + 1]);
      h[q * 4 + 2] = fmaf(xv, w.z, h[q * 4 + 2]);
      h[q * 4 + 3] = fmaf(xv, w.w, h[q * 4 + 3]);
    }
  }
  float y[AF];
#pragma unroll
  for (int f = 0; f < AF; ++f) y[f] = 0.0f;
#pragma unroll
  for (int j = 0; j < 32; ++j) {
    const float hv = fmaxf(h[j], 0.0f);
    const float4* wrow = (const float4*)(Wb + j * AF);
#pragma unroll
    for (int q = 0; q < 16; ++q) {
      const float4 w = wrow[q];
      y[q * 4 + 0] = fmaf(hv, w.x, y[q * 4 + 0]);
      y[q * 4 + 1] = fmaf(hv, w.y, y[q * 4 + 1]);
      y[q * 4 + 2] = fmaf(hv, w.z, y[q * 4 + 2]);
      y[q * 4 + 3] = fmaf(hv, w.w, y[q * 4 + 3]);
    }
  }
#pragma unroll
  for (int f = 0; f < AF; ++f) x[f] = (x[f] + fmaxf(y[f], 0.0f)) * kInvSqrt2;
}

__global__ __launch_bounds__(256) void k_node_mlp(
    const float* __restrict__ ns, const int* __restrict__ caidx,
    const float* __restrict__ nmean, const float* __restrict__ ninv,
    const float* __restrict__ g2, const float* __restrict__ b2,
    const float* __restrict__ Wr1a, const float* __restrict__ Wr1b,
    const float* __restrict__ Wr2a, const float* __restrict__ Wr2b,
    const float* __restrict__ atom_fea, float* __restrict__ out)
{
  __shared__ alignas(16) float sA1[AF * 32], sB1[32 * AF];
  __shared__ alignas(16) float sA2[AF * 32], sB2[32 * AF];  // 32 KB total
  for (int i = threadIdx.x; i < AF * 32; i += 256) {
    sA1[i] = Wr1a[i]; sB1[i] = Wr1b[i]; sA2[i] = Wr2a[i]; sB2[i] = Wr2b[i];
  }
  __syncthreads();
  for (int n = blockIdx.x * 256 + threadIdx.x; n < kNodes; n += gridDim.x * 256) {
    const int c = caidx[n];
    const float* xp = ns    + (size_t)n * AF;
    const float* mu = nmean + (size_t)c * AF;
    const float* iv = ninv  + (size_t)c * AF;
    float x[AF];
#pragma unroll
    for (int f = 0; f < AF; ++f) x[f] = (xp[f] - mu[f]) * iv[f] * g2[f] + b2[f];
    mlp_block(x, sA1, sB1);
    mlp_block(x, sA2, sB2);
    const float* af = atom_fea + (size_t)n * AF;
    float* op = out + (size_t)n * AF;
#pragma unroll
    for (int f = 0; f < AF; ++f) op[f] = kInvSqrt2 * fmaxf(af[f] + x[f], 0.0f);
  }
}

// ---------------- host launcher ----------------
extern "C" void kernel_launch(void* const* d_in, const int* in_sizes, int n_in,
                              void* d_out, int out_size, void* d_ws, size_t ws_size,
                              hipStream_t stream)
{
  const float* atom_fea = (const float*)d_in[0];
  const float* edge     = (const float*)d_in[1];
  const float* rbf      = (const float*)d_in[2];
  const int*   nbr_idx  = (const int*)d_in[3];
  const int*   caidx    = (const int*)d_in[4];
  const int*   ceidx    = (const int*)d_in[5];
  const float* W_rbf    = (const float*)d_in[6];
  const float* W_full   = (const float*)d_in[7];
  const float* W_mask   = (const float*)d_in[8];
  const float* g1       = (const float*)d_in[9];
  const float* b1       = (const float*)d_in[10];
  const float* g2       = (const float*)d_in[11];
  const float* b2       = (const float*)d_in[12];
  const float* Wr1a     = (const float*)d_in[13];
  const float* Wr1b     = (const float*)d_in[14];
  const float* Wr2a     = (const float*)d_in[15];
  const float* Wr2b     = (const float*)d_in[16];

  float* ws  = (float*)d_ws;
  float* out = (float*)d_out;

  float* gated    = ws + O_GATED;
  float* esum     = ws + O_ESUM;
  float* esumsq   = ws + O_ESUMSQ;
  float* ecnt     = ws + O_ECNT;
  float* node_sum = ws + O_NODESUM;
  float* node_cnt = ws + O_NODECNT;
  float* nsum     = ws + O_NSUM;
  float* nsumsq   = ws + O_NSUMSQ;
  float* ncnt     = ws + O_NCNT;
  float* emean    = ws + O_EMEAN;
  float* einv     = ws + O_EINV;
  float* nsbuf    = ws + O_NS;
  float* nmean    = ws + O_NMEAN;
  float* ninv     = ws + O_NINV;

  // 0: zero all accumulator regions (ws is poisoned, not re-zeroed by harness)
  k_zero<<<4096, 256, 0, stream>>>(ws + O_ZERO_BEG, O_ZERO_END - O_ZERO_BEG);

  // 1: big edge GEMM via WMMA bf16 -> gated
  k_edge_gemm<<<2500, 64, 0, stream>>>(atom_fea, edge, rbf, nbr_idx, W_rbf, W_full, gated);

  // 2: per-crystal edge stats (one pass sum & sumsq) + counts
  k_edge_stats<<<dim3(240, 4, 1), 256, 0, stream>>>(gated, ceidx, esum, esumsq);
  k_edge_cnt<<<240, 256, 0, stream>>>(ceidx, ecnt);
  k_finalize<<<(kCrys * NOUT + 255) / 256, 256, 0, stream>>>(
      esum, esumsq, ecnt, emean, einv, NOUT, kCrys * NOUT);

  // 3: normalize + gate + scatter messages to node accumulators
  k_edge_msg<<<(kEdges + 255) / 256, 256, 0, stream>>>(
      gated, ceidx, nbr_idx, emean, einv, g1, b1, W_mask, node_sum, node_cnt);

  // 4: node mean-pool + node crystal stats
  k_node_reduce<<<(kNodes + 255) / 256, 256, 0, stream>>>(
      node_sum, node_cnt, caidx, nsbuf, nsum, nsumsq, ncnt);
  k_finalize<<<(kCrys * AF + 255) / 256, 256, 0, stream>>>(
      nsum, nsumsq, ncnt, nmean, ninv, AF, kCrys * AF);

  // 5: node norm + residual MLPs + final relu/scale
  k_node_mlp<<<(kNodes + 255) / 256, 256, 0, stream>>>(
      nsbuf, caidx, nmean, ninv, g2, b2, Wr1a, Wr1b, Wr2a, Wr2b, atom_fea, out);

  (void)in_sizes; (void)n_in; (void)out_size; (void)ws_size;
}
